// GraphAttentionLayer_721554506536
// MI455X (gfx1250) — matrix operand order repaired and checked by reading, not compiled, not run on hardware
//
#include <hip/hip_runtime.h>
#include <hip/hip_bf16.h>

// ---------------------------------------------------------------------------
// GraphAttentionLayer for MI455X (gfx1250, wave32, WMMA f16 16x16x32)
//   out = 0.5 * softmax(cos(Xn,Yn) masked by adj) @ Y + 0.5 * X
// Cosine logits are bounded in [-1,1] -> fixed-shift softmax exp(s-1):
// no running max, no accumulator rescale, no LDS transpose.
// S is computed transposed (Yn * Xn^T) so exp'd P lands directly in the
// A-fragment register layout of the P@Y WMMA. Flash-decoding column split
// for occupancy; partials merged by a tiny combine kernel.
// ---------------------------------------------------------------------------

typedef __attribute__((ext_vector_type(16))) _Float16 v16h;
typedef __attribute__((ext_vector_type(8)))  float    v8f;
typedef __attribute__((ext_vector_type(4)))  float    v4f;

#define DDIM 256

union Frag {
    v16h     v;
    uint4    u[2];
    _Float16 h[16];
};

union H8 {
    uint4    u;
    _Float16 h[8];
};

__device__ __forceinline__ v8f wmma_f16(v16h a, v16h b, v8f c) {
    // D(16x16,f32) = A(16x32,f16) x B(32x16,f16) + C
    return __builtin_amdgcn_wmma_f32_16x16x32_f16(
        /*neg_a=*/false, a, /*neg_b=*/false, b,
        /*c_mod=*/(short)0, c, /*reuse_a=*/false, /*reuse_b=*/false);
}

// A-operand fragment (16xK rows of the M-dim operand), ISA 16-bit A layout:
//   lane l: row = l&15, K-chunks at  hi*8  and  16 + hi*8   (hi = l>>4)
__device__ __forceinline__ v16h load_fragA(const _Float16* __restrict__ base,
                                           long rowStride, int lane) {
    const int row = lane & 15;
    const int o   = (lane >> 4) * 8;
    const _Float16* p = base + (long)row * rowStride;
    Frag f;
    f.u[0] = *reinterpret_cast<const uint4*>(p + o);
    f.u[1] = *reinterpret_cast<const uint4*>(p + o + 16);
    return f.v;
}

// B-operand fragment (Kx16, element (k,n) read as src[n*rowStride + k]),
// ISA 16-bit B layout: lane l: col n = l&15, K-chunks at hi*8 and 16 + hi*8
__device__ __forceinline__ v16h load_fragB(const _Float16* __restrict__ base,
                                           long rowStride, int lane) {
    const int col = lane & 15;
    const int o   = (lane >> 4) * 8;
    const _Float16* p = base + (long)col * rowStride;
    Frag f;
    f.u[0] = *reinterpret_cast<const uint4*>(p + o);
    f.u[1] = *reinterpret_cast<const uint4*>(p + o + 16);
    return f.v;
}

// ---------------------------------------------------------------------------
// Prep: row L2-normalize x -> f16
// ---------------------------------------------------------------------------
__global__ __launch_bounds__(256) void gat_prep_x(const float* __restrict__ x,
                                                  _Float16* __restrict__ xn,
                                                  int N) {
    const int row  = blockIdx.x * 8 + (threadIdx.x >> 5);
    const int lane = threadIdx.x & 31;
    const float* xr = x + (long)row * DDIM + lane * 8;
    float v[8];
    const float4 f0 = *reinterpret_cast<const float4*>(xr);
    const float4 f1 = *reinterpret_cast<const float4*>(xr + 4);
    v[0]=f0.x; v[1]=f0.y; v[2]=f0.z; v[3]=f0.w;
    v[4]=f1.x; v[5]=f1.y; v[6]=f1.z; v[7]=f1.w;
    float ss = 0.f;
#pragma unroll
    for (int i = 0; i < 8; ++i) ss += v[i] * v[i];
#pragma unroll
    for (int off = 16; off; off >>= 1) ss += __shfl_xor(ss, off, 32);
    const float inv = 1.0f / fmaxf(sqrtf(ss), 1e-8f);
    H8 h;
#pragma unroll
    for (int i = 0; i < 8; ++i) h.h[i] = (_Float16)(v[i] * inv);
    *reinterpret_cast<uint4*>(xn + (long)row * DDIM + lane * 8) = h.u;
}

// Prep: row L2-normalize y -> yn (f16); also raw y -> yT (f16, [D, M])
__global__ __launch_bounds__(256) void gat_prep_y(const float* __restrict__ y,
                                                  _Float16* __restrict__ yn,
                                                  _Float16* __restrict__ yT,
                                                  int M) {
    const int row  = blockIdx.x * 8 + (threadIdx.x >> 5);
    const int lane = threadIdx.x & 31;
    const float* yr = y + (long)row * DDIM + lane * 8;
    float v[8];
    const float4 f0 = *reinterpret_cast<const float4*>(yr);
    const float4 f1 = *reinterpret_cast<const float4*>(yr + 4);
    v[0]=f0.x; v[1]=f0.y; v[2]=f0.z; v[3]=f0.w;
    v[4]=f1.x; v[5]=f1.y; v[6]=f1.z; v[7]=f1.w;
    float ss = 0.f;
#pragma unroll
    for (int i = 0; i < 8; ++i) ss += v[i] * v[i];
#pragma unroll
    for (int off = 16; off; off >>= 1) ss += __shfl_xor(ss, off, 32);
    const float inv = 1.0f / fmaxf(sqrtf(ss), 1e-8f);
    H8 h;
#pragma unroll
    for (int i = 0; i < 8; ++i) h.h[i] = (_Float16)(v[i] * inv);
    *reinterpret_cast<uint4*>(yn + (long)row * DDIM + lane * 8) = h.u;
#pragma unroll
    for (int i = 0; i < 8; ++i)
        yT[(long)(lane * 8 + i) * M + row] = (_Float16)v[i];
}

// ---------------------------------------------------------------------------
// Fused attention: one wave per 16-row strip, streaming one M-chunk.
// Writes unnormalized partial O [cs,N,D] and row-sums l [cs,N].
// ---------------------------------------------------------------------------
__global__ __launch_bounds__(128) void gat_attn(
    const _Float16* __restrict__ xn,   // [N, D]
    const _Float16* __restrict__ yn,   // [M, D]
    const _Float16* __restrict__ yT,   // [D, M]
    const float*    __restrict__ adj,  // [N, M]
    float*          __restrict__ Opart,  // [cs, N, D]
    float*          __restrict__ lpart,  // [cs, N]
    int N, int M, int cs)
{
    const int wave  = threadIdx.x >> 5;
    const int lane  = threadIdx.x & 31;
    const int hi    = lane >> 4;
    const int col16 = lane & 15;
    const int rbase = blockIdx.x * 64 + wave * 16;
    const int chunk = blockIdx.y;
    const int jlen  = M / cs;
    const int j0    = chunk * jlen;

    // Resident Xn fragments (B-operand of S^T = Yn * Xn^T), K = 256.
    v16h xb[8];
#pragma unroll
    for (int kc = 0; kc < 8; ++kc)
        xb[kc] = load_fragB(xn + (long)rbase * DDIM + kc * 32, DDIM, lane);

    // O accumulators: C layout (x-row in VGPR index, d-col in lane).
    v8f o[16];
#pragma unroll
    for (int dt = 0; dt < 16; ++dt)
#pragma unroll
        for (int e = 0; e < 8; ++e) o[dt][e] = 0.f;

    float lacc = 0.f;   // per-lane partial softmax denominator (row = col16)

    // This lane's adj row, offset so element r of tile t is at +t*16+r.
    const float* arow = adj + (long)(rbase + col16) * M + hi * 8;

    const int njt = jlen >> 5;   // 32 columns per iteration
    for (int it = 0; it < njt; ++it) {
        const int jj = j0 + it * 32;

        // --- adjacency: 8 consecutive j per lane per tile (non-temporal) ---
        const v4f q00 = __builtin_nontemporal_load(
            reinterpret_cast<const v4f*>(arow + jj));
        const v4f q01 = __builtin_nontemporal_load(
            reinterpret_cast<const v4f*>(arow + jj + 4));
        const v4f q10 = __builtin_nontemporal_load(
            reinterpret_cast<const v4f*>(arow + jj + 16));
        const v4f q11 = __builtin_nontemporal_load(
            reinterpret_cast<const v4f*>(arow + jj + 20));
        float a0[8], a1[8];
#pragma unroll
        for (int i = 0; i < 4; ++i) {
            a0[i]     = q00[i];
            a0[i + 4] = q01[i];
            a1[i]     = q10[i];
            a1[i + 4] = q11[i];
        }
        __builtin_prefetch(arow + jj + 32, 0, 0);

        // --- S^T tiles: C layout has j in VGPR index, x-row in lane ---
        v8f s0, s1;
#pragma unroll
        for (int e = 0; e < 8; ++e) { s0[e] = 0.f; s1[e] = 0.f; }
#pragma unroll
        for (int kc = 0; kc < 8; ++kc) {
            const v16h ya0 = load_fragA(yn + (long)jj        * DDIM + kc * 32, DDIM, lane);
            const v16h ya1 = load_fragA(yn + (long)(jj + 16) * DDIM + kc * 32, DDIM, lane);
            s0 = wmma_f16(ya0, xb[kc], s0);
            s1 = wmma_f16(ya1, xb[kc], s1);
        }

        // --- fixed-shift masked softmax numerator; builds the P A-fragment
        //     directly in registers (t=0 -> K 8hi..8hi+7, t=1 -> 16+8hi..) ---
        Frag pf;
#pragma unroll
        for (int r = 0; r < 8; ++r) {
            const float p0 = (a0[r] > 0.5f) ? __expf(s0[r] - 1.0f) : 0.f;
            const float p1 = (a1[r] > 0.5f) ? __expf(s1[r] - 1.0f) : 0.f;
            lacc += p0 + p1;
            pf.h[r]     = (_Float16)p0;
            pf.h[r + 8] = (_Float16)p1;
        }

        // --- O += P(16x32) @ Y(32 x 16-per-d-tile) ---
#pragma unroll
        for (int dt = 0; dt < 16; ++dt) {
            const v16h yb = load_fragB(yT + (long)(dt * 16) * M + jj, M, lane);
            o[dt] = wmma_f16(pf.v, yb, o[dt]);
        }
    }

    // Lane l and l+16 hold disjoint j-subsets of the same x-row: one xor-16.
    const float ltot = lacc + __shfl_xor(lacc, 16, 32);

    // --- write partials (unnormalized O; per-row l) ---
#pragma unroll
    for (int dt = 0; dt < 16; ++dt)
#pragma unroll
        for (int r = 0; r < 8; ++r)
            __builtin_nontemporal_store(
                o[dt][r],
                Opart + ((long)chunk * N + rbase + r + 8 * hi) * DDIM
                      + dt * 16 + col16);
    if (hi == 0)
        lpart[(long)chunk * N + rbase + col16] = ltot;
}

// ---------------------------------------------------------------------------
// Merge column-split partials: out = 0.5 * (sum O_i) / (sum l_i) + 0.5 * x
// (fixed softmax shift -> no exp rebasing needed)
// ---------------------------------------------------------------------------
__global__ __launch_bounds__(256) void gat_combine(
    const float* __restrict__ Opart, const float* __restrict__ lpart,
    const float* __restrict__ x, float* __restrict__ out, int N, int cs)
{
    const long idx  = (long)blockIdx.x * blockDim.x + threadIdx.x;  // N*D total
    const long nrow = idx >> 8;
    const int  dcol = (int)(idx & 255);
    float denom = 0.f, val = 0.f;
    for (int i = 0; i < cs; ++i) {
        const long r = (long)i * N + nrow;
        denom += lpart[r];
        val   += Opart[r * DDIM + dcol];
    }
    out[idx] = 0.5f * (val / denom) + 0.5f * x[idx];
}

// ---------------------------------------------------------------------------
extern "C" void kernel_launch(void* const* d_in, const int* in_sizes, int n_in,
                              void* d_out, int out_size, void* d_ws, size_t ws_size,
                              hipStream_t stream) {
    const float* x   = (const float*)d_in[0];
    const float* y   = (const float*)d_in[1];
    const float* adj = (const float*)d_in[2];
    float* out = (float*)d_out;

    const int N = in_sizes[0] / DDIM;
    const int M = in_sizes[1] / DDIM;

    // workspace carve-up
    char* ws = (char*)d_ws;
    _Float16* xn = (_Float16*)ws;                   // N*D halfs
    _Float16* yn = xn + (long)N * DDIM;             // M*D halfs
    _Float16* yT = yn + (long)M * DDIM;             // D*M halfs
    size_t off = (size_t)((char*)(yT + (long)DDIM * M) - ws);
    off = (off + 255) & ~(size_t)255;

    // flash-decoding split: largest cs whose partials fit the workspace
    int cs = 8;
    while (cs > 1) {
        const size_t need = off + (size_t)cs * N * sizeof(float)
                                + (size_t)cs * N * DDIM * sizeof(float);
        if (need <= ws_size && (M % (cs * 32)) == 0) break;
        cs >>= 1;
    }
    float* lpart = (float*)(ws + off);
    float* Opart = lpart + (size_t)cs * N;

    gat_prep_x<<<N / 8, 256, 0, stream>>>(x, xn, N);
    gat_prep_y<<<M / 8, 256, 0, stream>>>(y, yn, yT, M);

    dim3 grid(N / 64, cs);
    gat_attn<<<grid, 128, 0, stream>>>(xn, yn, yT, adj, Opart, lpart, N, M, cs);

    gat_combine<<<(N * DDIM) / 256, 256, 0, stream>>>(Opart, lpart, x, out, N, cs);
}
